// QuantumNATExtendedQML_65481071409864
// MI455X (gfx1250) — matrix operand order repaired and checked by reading, not compiled, not run on hardware
//
#include <hip/hip_runtime.h>
#include <hip/hip_bf16.h>
#include <math.h>

typedef __attribute__((ext_vector_type(16))) _Float16 v16h;
typedef __attribute__((ext_vector_type(8)))  float    v8f;
typedef __attribute__((ext_vector_type(4)))  int      i4vec;

#define EPS_BN 1e-5f

// LDS tile geometry for conv2: 8 ic x 6 input rows x 120 cols (f16).
// Valid x lives at [LPAD, LPAD+112); LPAD=8 keeps every staged 8-half chunk
// 16-byte aligned so staging can use b128 ops / async-to-LDS.
// A 480-half zero region sits at [TSIZE, TSIZE+ZPAD): for padding K >= 72 the
// per-lane offset is TSIZE, so base(7..478) + TSIZE always reads zeros.
#define TROWS 6
#define TW    120
#define LPAD  8
#define TSIZE (8 * TROWS * TW)   // 5760 halves
#define ZPAD  480
#define SHALVES (TSIZE + ZPAD)   // 6240 halves = 12480 B (16B multiple)

#if defined(__has_builtin)
#  if __has_builtin(__builtin_amdgcn_global_load_async_to_lds_b128)
#    define HAVE_ASYNC_LDS 1
#  endif
#endif
#ifndef HAVE_ASYNC_LDS
#  define HAVE_ASYNC_LDS 0
#endif

#if HAVE_ASYNC_LDS
typedef __attribute__((address_space(1))) i4vec* as1_i4p;
typedef __attribute__((address_space(3))) i4vec* as3_i4p;
#endif

// ---------------------------------------------------------------------------
// Prep: zero stats accumulators and build the conv2 weight A-fragment
// (16 oc x 96 K, f16, K padded 72->96) in a contiguous-K A layout (not the ISA table order):
//   lane = M + 16*khalf ; element e -> K = 16*khalf + e
// ---------------------------------------------------------------------------
__global__ __launch_bounds__(256) void k_prep(const float* __restrict__ w2,
                                              _Float16* __restrict__ afrag,
                                              float* __restrict__ stats) {
  int t = threadIdx.x;
  for (int i = t; i < 2096; i += 256) stats[i] = 0.0f;
  if (t < 96) {
    int c = t >> 5, lane = t & 31;
    int m = lane & 15, khalf = lane >> 4;
    #pragma unroll
    for (int e = 0; e < 16; ++e) {
      int K = 32 * c + 16 * khalf + e;
      _Float16 v = (_Float16)0.0f;
      if (K < 72) v = (_Float16)w2[m * 72 + K];   // w2[oc][ic*9+kh*3+kw]
      afrag[(c * 32 + lane) * 16 + e] = v;
    }
  }
}

// ---------------------------------------------------------------------------
// Pass 1a: conv1 stats (per-channel sum / sumsq of conv output incl. bias)
// ---------------------------------------------------------------------------
__global__ __launch_bounds__(256) void k_conv1_stats(
    const float* __restrict__ x, const float* __restrict__ w1,
    const float* __restrict__ b1, float* __restrict__ sum8,
    float* __restrict__ sq8) {
  __shared__ float ssum[8], ssq[8];
  int t = threadIdx.x;
  if (t < 8) { ssum[t] = 0.0f; ssq[t] = 0.0f; }
  __syncthreads();

  long idx = (long)blockIdx.x * 256 + t;        // 128*224*224 pixels
  int px = (int)(idx % 224);
  long r = idx / 224;
  int py = (int)(r % 224);
  int b  = (int)(r / 224);
  const float* xb = x + (long)b * 224 * 224;

  float patch[3][3];
  #pragma unroll
  for (int dy = 0; dy < 3; ++dy)
    #pragma unroll
    for (int dx = 0; dx < 3; ++dx) {
      int yy = py + dy - 1, xx = px + dx - 1;
      patch[dy][dx] = (yy >= 0 && yy < 224 && xx >= 0 && xx < 224)
                          ? xb[yy * 224 + xx] : 0.0f;
    }

  #pragma unroll
  for (int oc = 0; oc < 8; ++oc) {
    float acc = b1[oc];
    #pragma unroll
    for (int dy = 0; dy < 3; ++dy)
      #pragma unroll
      for (int dx = 0; dx < 3; ++dx)
        acc = fmaf(w1[oc * 9 + dy * 3 + dx], patch[dy][dx], acc);
    float s = acc, q = acc * acc;
    #pragma unroll
    for (int mm = 16; mm >= 1; mm >>= 1) {
      s += __shfl_xor(s, mm, 32);
      q += __shfl_xor(q, mm, 32);
    }
    if ((t & 31) == 0) { atomicAdd(&ssum[oc], s); atomicAdd(&ssq[oc], q); }
  }
  __syncthreads();
  if (t < 8) { atomicAdd(&sum8[t], ssum[t]); atomicAdd(&sq8[t], ssq[t]); }
}

// ---------------------------------------------------------------------------
// Pass 1b: recompute conv1, apply BN+ReLU, 2x2 maxpool, write h1 as f16
// ---------------------------------------------------------------------------
__global__ __launch_bounds__(256) void k_conv1_apply(
    const float* __restrict__ x, const float* __restrict__ w1,
    const float* __restrict__ b1, const float* __restrict__ g1,
    const float* __restrict__ bb1, const float* __restrict__ sum8,
    const float* __restrict__ sq8, _Float16* __restrict__ h1) {
  long idx = (long)blockIdx.x * 256 + threadIdx.x;   // 128*112*112
  int px = (int)(idx % 112);
  long r = idx / 112;
  int py = (int)(r % 112);
  int b  = (int)(r / 112);
  const float* xb = x + (long)b * 224 * 224;

  float patch[4][4];
  int ybase = py * 2 - 1, xbase = px * 2 - 1;
  #pragma unroll
  for (int dy = 0; dy < 4; ++dy)
    #pragma unroll
    for (int dx = 0; dx < 4; ++dx) {
      int yy = ybase + dy, xx = xbase + dx;
      patch[dy][dx] = (yy >= 0 && yy < 224 && xx >= 0 && xx < 224)
                          ? xb[yy * 224 + xx] : 0.0f;
    }

  const float Np = 128.0f * 224.0f * 224.0f;
  #pragma unroll
  for (int oc = 0; oc < 8; ++oc) {
    float mean  = sum8[oc] / Np;
    float var   = sq8[oc] / Np - mean * mean;
    float scale = g1[oc] * rsqrtf(var + EPS_BN);
    float shift = bb1[oc] - mean * scale;
    float bias  = b1[oc];
    float best  = 0.0f;  // ReLU floor
    #pragma unroll
    for (int dy = 0; dy < 2; ++dy)
      #pragma unroll
      for (int dx = 0; dx < 2; ++dx) {
        float acc = bias;
        #pragma unroll
        for (int kh = 0; kh < 3; ++kh)
          #pragma unroll
          for (int kw = 0; kw < 3; ++kw)
            acc = fmaf(w1[oc * 9 + kh * 3 + kw], patch[dy + kh][dx + kw], acc);
        best = fmaxf(best, acc * scale + shift);
      }
    h1[(((long)b * 8 + oc) * 112 + py) * 112 + px] = (_Float16)best;
  }
}

// ---------------------------------------------------------------------------
// conv2 helpers: LDS-staged implicit GEMM tile.
// B layout (contiguous K, same as afrag): lane = N + 16*khalf, element e -> K = 32c+16*khalf+e
// LDS offset of element K at (output row yr, column x0+n):
//   base + OFF[c][e], base = yr*TW + x0 + n + (LPAD-1).
// OFF is selected once per lane (khalf) from compile-time deltas; K >= 72 maps
// to TSIZE so base+OFF reads the zero region for every base in [7, 478].
// ---------------------------------------------------------------------------
__device__ constexpr int b_delta(int K) {
  return (K < 72)
             ? ((K / 9) * TROWS + ((K % 9) / 3)) * TW + ((K % 9) % 3)
             : TSIZE;
}

__device__ __forceinline__ void conv2_offsets(int khalf, int off[3][16]) {
  #pragma unroll
  for (int c = 0; c < 3; ++c)
    #pragma unroll
    for (int e = 0; e < 16; ++e) {
      const int dLo = b_delta(32 * c + e);        // khalf == 0
      const int dHi = b_delta(32 * c + 16 + e);   // khalf == 1
      off[c][e] = khalf ? dHi : dLo;              // one cndmask, hoisted
    }
}

__device__ __forceinline__ v8f conv2_row(const _Float16* S,
                                         const v16h* __restrict__ afrag,
                                         int base, int lane,
                                         const int off[3][16]) {
  v8f acc = {};
  #pragma unroll
  for (int c = 0; c < 3; ++c) {
    v16h bf;
    #pragma unroll
    for (int e = 0; e < 16; ++e)
      bf[e] = S[base + off[c][e]];   // one v_add + one ds_load_u16
    v16h a = afrag[c * 32 + lane];
    acc = __builtin_amdgcn_wmma_f32_16x16x32_f16(
        false, a, false, bf, (short)0, acc, false, false);
  }
  return acc;
}

// Cooperative LDS stage: vectorized zero fill, then 16B-aligned b128 copies
// (async-to-LDS on toolchains that expose the builtin).
__device__ __forceinline__ void conv2_stage(_Float16* S,
                                            const _Float16* __restrict__ h1,
                                            int b, int y0, int t) {
  uint4 z4; z4.x = 0u; z4.y = 0u; z4.z = 0u; z4.w = 0u;
  uint4* S4 = (uint4*)S;
  for (int i = t; i < SHALVES / 8; i += 224) S4[i] = z4;
  __syncthreads();
  for (int j = t; j < 8 * TROWS * 14; j += 224) {   // 672 chunks of 8 halves
    int xx = (j % 14) * 8;
    int rem = j / 14;
    int rr = rem % TROWS;
    int ic = rem / TROWS;
    int yy = y0 - 1 + rr;
    if (yy >= 0 && yy < 112) {
      _Float16* dst = &S[(ic * TROWS + rr) * TW + LPAD + xx];
      const _Float16* src = &h1[(((long)b * 8 + ic) * 112 + yy) * 112 + xx];
#if HAVE_ASYNC_LDS
      __builtin_amdgcn_global_load_async_to_lds_b128(
          (as1_i4p)(i4vec*)(void*)(const void*)src,
          (as3_i4p)(i4vec*)(void*)dst, 0, 0);
#else
      *(uint4*)dst = *(const uint4*)src;
#endif
    }
  }
#if HAVE_ASYNC_LDS
#  if __has_builtin(__builtin_amdgcn_s_wait_asynccnt)
  __builtin_amdgcn_s_wait_asynccnt(0);
#  else
  asm volatile("s_wait_asynccnt 0x0" ::: "memory");
#  endif
#endif
  __syncthreads();
}

// ---------------------------------------------------------------------------
// Pass 2a: conv2 stats (WMMA). Block = 7 waves, one (image, 4-row band).
// ---------------------------------------------------------------------------
__global__ __launch_bounds__(224) void k_conv2_stats(
    const _Float16* __restrict__ h1, const v16h* __restrict__ afrag,
    const float* __restrict__ b2, float* __restrict__ sum16,
    float* __restrict__ sq16) {
  __shared__ _Float16 S[SHALVES];
  __shared__ float ssum[16], ssq[16];
  int t = threadIdx.x;
  if (t < 16) { ssum[t] = 0.0f; ssq[t] = 0.0f; }

  int wave = t >> 5, lane = t & 31;
  int b = blockIdx.y, y0 = blockIdx.x * 4, x0 = wave * 16;
  conv2_stage(S, h1, b, y0, t);

  int n = lane & 15;
  int off[3][16];
  conv2_offsets(lane >> 4, off);
  v8f d[4];
  #pragma unroll
  for (int yr = 0; yr < 4; ++yr)
    d[yr] = conv2_row(S, afrag, yr * TW + x0 + n + (LPAD - 1), lane, off);

  int ocbase = (lane >> 4) * 8;
  #pragma unroll
  for (int r = 0; r < 8; ++r) {
    float bias = b2[ocbase + r];
    float s = 0.0f, q = 0.0f;
    #pragma unroll
    for (int yr = 0; yr < 4; ++yr) {
      float v = d[yr][r] + bias;
      s += v;
      q += v * v;
    }
    #pragma unroll
    for (int mm = 8; mm >= 1; mm >>= 1) {   // reduce within 16-lane half
      s += __shfl_xor(s, mm, 32);
      q += __shfl_xor(q, mm, 32);
    }
    if ((lane & 15) == 0) {
      atomicAdd(&ssum[ocbase + r], s);
      atomicAdd(&ssq[ocbase + r], q);
    }
  }
  __syncthreads();
  if (t < 16) { atomicAdd(&sum16[t], ssum[t]); atomicAdd(&sq16[t], ssq[t]); }
}

// ---------------------------------------------------------------------------
// Pass 2b: recompute conv2 (WMMA), BN+ReLU, 2x2 maxpool, accumulate per-(b,oc)
// spatial sums into featsum.
// ---------------------------------------------------------------------------
__global__ __launch_bounds__(224) void k_conv2_apply(
    const _Float16* __restrict__ h1, const v16h* __restrict__ afrag,
    const float* __restrict__ b2, const float* __restrict__ g2,
    const float* __restrict__ bb2, const float* __restrict__ sum16,
    const float* __restrict__ sq16, float* __restrict__ featsum) {
  __shared__ _Float16 S[SHALVES];
  __shared__ float sfeat[16];
  int t = threadIdx.x;
  if (t < 16) sfeat[t] = 0.0f;

  int wave = t >> 5, lane = t & 31;
  int b = blockIdx.y, y0 = blockIdx.x * 4, x0 = wave * 16;
  conv2_stage(S, h1, b, y0, t);

  int n = lane & 15;
  int off[3][16];
  conv2_offsets(lane >> 4, off);
  v8f d[4];
  #pragma unroll
  for (int yr = 0; yr < 4; ++yr)
    d[yr] = conv2_row(S, afrag, yr * TW + x0 + n + (LPAD - 1), lane, off);

  const float Np = 128.0f * 112.0f * 112.0f;
  int ocbase = (lane >> 4) * 8;
  #pragma unroll
  for (int r = 0; r < 8; ++r) {
    int oc = ocbase + r;
    float mean  = sum16[oc] / Np;
    float var   = sq16[oc] / Np - mean * mean;
    float scale = g2[oc] * rsqrtf(var + EPS_BN);
    float shift = bb2[oc] + (b2[oc] - mean) * scale;
    float v0 = fmaxf(d[0][r] * scale + shift, 0.0f);
    float v1 = fmaxf(d[1][r] * scale + shift, 0.0f);
    float v2 = fmaxf(d[2][r] * scale + shift, 0.0f);
    float v3 = fmaxf(d[3][r] * scale + shift, 0.0f);
    float p01 = fmaxf(v0, v1);                    // pool over row pair 0
    float p23 = fmaxf(v2, v3);                    // pool over row pair 1
    p01 = fmaxf(p01, __shfl_xor(p01, 1, 32));     // pool over x pair
    p23 = fmaxf(p23, __shfl_xor(p23, 1, 32));
    float contrib = ((lane & 1) == 0) ? (p01 + p23) : 0.0f;
    #pragma unroll
    for (int mm = 8; mm >= 1; mm >>= 1)
      contrib += __shfl_xor(contrib, mm, 32);
    if ((lane & 15) == 0) atomicAdd(&sfeat[oc], contrib);
  }
  __syncthreads();
  if (t < 16) atomicAdd(&featsum[(long)b * 16 + t], sfeat[t]);
}

// ---------------------------------------------------------------------------
// 4-qubit statevector gates. Wire q maps to bit (3-q): mask = 8 >> q.
// ---------------------------------------------------------------------------
__device__ __forceinline__ void gate_ry(float* re, float* im, int q, float t) {
  float c = cosf(0.5f * t), s = sinf(0.5f * t);
  int m = 8 >> q;
  #pragma unroll
  for (int a = 0; a < 16; ++a) {
    if (a & m) continue;
    int bb = a | m;
    float r0 = re[a], i0 = im[a], r1 = re[bb], i1 = im[bb];
    re[a]  = c * r0 - s * r1;  im[a]  = c * i0 - s * i1;
    re[bb] = s * r0 + c * r1;  im[bb] = s * i0 + c * i1;
  }
}

__device__ __forceinline__ void gate_rz(float* re, float* im, int q, float t) {
  float c = cosf(0.5f * t), s = sinf(0.5f * t);
  int m = 8 >> q;
  #pragma unroll
  for (int a = 0; a < 16; ++a) {
    float r = re[a], i = im[a];
    if (a & m) { re[a] = r * c - i * s; im[a] = i * c + r * s; }   // e^{+it/2}
    else       { re[a] = r * c + i * s; im[a] = i * c - r * s; }   // e^{-it/2}
  }
}

__device__ __forceinline__ void gate_rx(float* re, float* im, int q, float t) {
  float c = cosf(0.5f * t), s = sinf(0.5f * t);
  int m = 8 >> q;
  #pragma unroll
  for (int a = 0; a < 16; ++a) {
    if (a & m) continue;
    int bb = a | m;
    float r0 = re[a], i0 = im[a], r1 = re[bb], i1 = im[bb];
    re[a]  = c * r0 + s * i1;  im[a]  = c * i0 - s * r1;
    re[bb] = c * r1 + s * i0;  im[bb] = c * i1 - s * r0;
  }
}

__device__ __forceinline__ void gate_cnot(float* re, float* im, int c, int t) {
  int mc = 8 >> c, mt = 8 >> t;
  #pragma unroll
  for (int a = 0; a < 16; ++a) {
    if ((a & mc) && !(a & mt)) {
      int bb = a | mt;
      float tr = re[a]; re[a] = re[bb]; re[bb] = tr;
      float ti = im[a]; im[a] = im[bb]; im[bb] = ti;
    }
  }
}

// ---------------------------------------------------------------------------
// Finalize: feat normalize -> quantum circuit -> batch norm. One block,
// one thread per batch element.
// ---------------------------------------------------------------------------
__global__ __launch_bounds__(128) void k_finalize(
    const float* __restrict__ featsum, const float* __restrict__ theta,
    const float* __restrict__ rho, const float* __restrict__ ng,
    const float* __restrict__ nb, float* __restrict__ out) {
  __shared__ float rs[128], rq[128];
  __shared__ float cs[4], cq[4];
  int b = threadIdx.x;
  if (b < 4) { cs[b] = 0.0f; cq[b] = 0.0f; }

  float feat[4];
  #pragma unroll
  for (int i = 0; i < 4; ++i)
    feat[i] = featsum[b * 16 + i] * (1.0f / (56.0f * 56.0f));

  rs[b] = feat[0] + feat[1] + feat[2] + feat[3];
  rq[b] = feat[0] * feat[0] + feat[1] * feat[1] +
          feat[2] * feat[2] + feat[3] * feat[3];
  __syncthreads();
  for (int st = 64; st > 0; st >>= 1) {
    if (b < st) { rs[b] += rs[b + st]; rq[b] += rq[b + st]; }
    __syncthreads();
  }
  float mean = rs[0] * (1.0f / 512.0f);
  float var  = (rq[0] - 512.0f * mean * mean) * (1.0f / 511.0f);  // ddof=1
  float sd   = sqrtf(fmaxf(var, 0.0f));
  const float kPi = 3.14159265358979323846f;

  float scaled[4];
  #pragma unroll
  for (int i = 0; i < 4; ++i)
    scaled[i] = (feat[i] - mean) / (sd + 1e-6f) * kPi;

  float re[16], im[16];
  #pragma unroll
  for (int a = 0; a < 16; ++a) { re[a] = 0.0f; im[a] = 0.0f; }
  re[0] = 1.0f;

  #pragma unroll
  for (int i = 0; i < 4; ++i) gate_ry(re, im, i, scaled[i]);
  #pragma unroll
  for (int i = 0; i < 4; ++i) {
    gate_ry(re, im, i, theta[i * 3 + 0]);
    gate_rz(re, im, i, theta[i * 3 + 1]);
    gate_rx(re, im, i, theta[i * 3 + 2]);
  }
  gate_cnot(re, im, 0, 1); gate_cnot(re, im, 1, 2); gate_cnot(re, im, 2, 3);
  #pragma unroll
  for (int i = 0; i < 4; ++i) {
    gate_ry(re, im, i, rho[i * 3 + 0]);
    gate_rz(re, im, i, rho[i * 3 + 1]);
    gate_rx(re, im, i, rho[i * 3 + 2]);
  }
  gate_cnot(re, im, 1, 0); gate_cnot(re, im, 2, 1); gate_cnot(re, im, 3, 2);

  float ev[4];
  #pragma unroll
  for (int qd = 0; qd < 4; ++qd) {
    int m = 8 >> qd;
    float e = 0.0f;
    #pragma unroll
    for (int a = 0; a < 16; ++a) {
      float p = re[a] * re[a] + im[a] * im[a];
      e += (a & m) ? -p : p;
    }
    ev[qd] = e;
  }

  #pragma unroll
  for (int i = 0; i < 4; ++i) {
    atomicAdd(&cs[i], ev[i]);
    atomicAdd(&cq[i], ev[i] * ev[i]);
  }
  __syncthreads();
  #pragma unroll
  for (int i = 0; i < 4; ++i) {
    float mu = cs[i] * (1.0f / 128.0f);
    float v  = cq[i] * (1.0f / 128.0f) - mu * mu;   // ddof=0
    out[b * 4 + i] = (ev[i] - mu) * rsqrtf(v + EPS_BN) * ng[i] + nb[i];
  }
}

// ---------------------------------------------------------------------------
extern "C" void kernel_launch(void* const* d_in, const int* in_sizes, int n_in,
                              void* d_out, int out_size, void* d_ws,
                              size_t ws_size, hipStream_t stream) {
  const float* x     = (const float*)d_in[0];
  const float* w1    = (const float*)d_in[1];
  const float* b1    = (const float*)d_in[2];
  const float* g1    = (const float*)d_in[3];
  const float* bb1   = (const float*)d_in[4];
  const float* w2    = (const float*)d_in[5];
  const float* b2    = (const float*)d_in[6];
  const float* g2    = (const float*)d_in[7];
  const float* bb2   = (const float*)d_in[8];
  const float* theta = (const float*)d_in[9];
  const float* rho   = (const float*)d_in[10];
  const float* ng    = (const float*)d_in[11];
  const float* nb    = (const float*)d_in[12];

  char* ws = (char*)d_ws;
  _Float16* afrag = (_Float16*)ws;            // 3 chunks * 32 lanes * 16 h = 3 KB
  float* stats    = (float*)(ws + 4096);      // 2096 floats, zeroed every launch
  float* bn1_sum  = stats;                    // 8
  float* bn1_sq   = stats + 8;                // 8
  float* bn2_sum  = stats + 16;               // 16
  float* bn2_sq   = stats + 32;               // 16
  float* featsum  = stats + 48;               // 128*16
  _Float16* h1    = (_Float16*)(ws + 16384);  // 128*8*112*112 f16 = 25.7 MB

  k_prep<<<1, 256, 0, stream>>>(w2, afrag, stats);
  k_conv1_stats<<<25088, 256, 0, stream>>>(x, w1, b1, bn1_sum, bn1_sq);
  k_conv1_apply<<<6272, 256, 0, stream>>>(x, w1, b1, g1, bb1, bn1_sum, bn1_sq, h1);
  dim3 grid2(28, 128);  // (4-row bands, batch)
  k_conv2_stats<<<grid2, 224, 0, stream>>>(h1, (const v16h*)afrag, b2, bn2_sum,
                                           bn2_sq);
  k_conv2_apply<<<grid2, 224, 0, stream>>>(h1, (const v16h*)afrag, b2, g2, bb2,
                                           bn2_sum, bn2_sq, featsum);
  k_finalize<<<1, 128, 0, stream>>>(featsum, theta, rho, ng, nb, (float*)d_out);

  (void)in_sizes; (void)n_in; (void)out_size; (void)ws_size;
}
